// DetectionTargetLayer_74320114090264
// MI455X (gfx1250) — compile-verified, matrix operand order checked by reading, not run on hardware
//
#include <hip/hip_runtime.h>
#include <hip/hip_bf16.h>

// DetectionTargetLayer for MI455X (gfx1250, wave32).
// Shapes fixed by setup_inputs(): B=16, N=2000 proposals, G=200 GT, T=200 slots,
// source masks 56x56, output mini-masks 28x28.
// Outputs concatenated flat in d_out (float32):
//   rois   [B,T,4]      at 0
//   cls    [B,T]        at B*T*4
//   deltas [B,T,4]      at B*T*5
//   masks  [B,T,28,28]  at B*T*9

#define BN   16
#define NP   2000
#define NG   200
#define TT   200
#define PMAX 66
#define HMS  56      // source mask H=W
#define MOx  28      // output mini-mask H=W
#define EPSF 1e-6f

typedef __attribute__((ext_vector_type(16))) _Float16 v16h;
typedef __attribute__((ext_vector_type(8)))  _Float16 h8;
typedef __attribute__((ext_vector_type(8)))  float    v8f;

__device__ __forceinline__ int imin(int a, int b) { return a < b ? a : b; }
__device__ __forceinline__ int imax(int a, int b) { return a > b ? a : b; }

// ---------------------------------------------------------------------------
// Kernel 1: per-proposal IoU reduction over 200 GT boxes.
// ---------------------------------------------------------------------------
__global__ void k_iou(const float* __restrict__ prop, const int* __restrict__ cls,
                      const float* __restrict__ gtb, int* __restrict__ assign,
                      int* __restrict__ flag) {
    int idx = blockIdx.x * blockDim.x + threadIdx.x;
    if (idx >= BN * NP) return;
    int b = idx / NP;
    const float* p = prop + (size_t)idx * 4;
    float ay1 = p[0], ax1 = p[1], ay2 = p[2], ax2 = p[3];
    bool valid_prop = (ay1 != 0.f) || (ax1 != 0.f) || (ay2 != 0.f) || (ax2 != 0.f);
    float area_a = (ay2 - ay1) * (ax2 - ax1);
    float best = -1e30f;      // first-max => jnp.argmax semantics
    int   bi = 0;
    float crowd_max = 0.f;
    for (int g = 0; g < NG; ++g) {
        const float* q = gtb + (size_t)(b * NG + g) * 4;
        float by1 = q[0], bx1 = q[1], by2 = q[2], bx2 = q[3];
        int c = cls[b * NG + g];
        bool valid_gt = (by1 != 0.f) || (bx1 != 0.f) || (by2 != 0.f) || (bx2 != 0.f);
        bool nc = (c > 0) && valid_gt;
        bool cr = (c < 0) && valid_gt;
        float iy = fmaxf(fminf(ay2, by2) - fmaxf(ay1, by1), 0.f);
        float ix = fmaxf(fminf(ax2, bx2) - fmaxf(ax1, bx1), 0.f);
        float inter = iy * ix;
        float uni = area_a + (by2 - by1) * (bx2 - bx1) - inter;
        float ov = inter / fmaxf(uni, EPSF);
        float ovnc = nc ? ov : -1.f;
        if (ovnc > best) { best = ovnc; bi = g; }
        if (cr) crowd_max = fmaxf(crowd_max, ov);
    }
    assign[idx] = bi;
    int f = 0;
    if (valid_prop && best >= 0.5f) f = 1;
    else if (valid_prop && best < 0.5f && crowd_max < 0.001f) f = 2;
    flag[idx] = f;
}

// ---------------------------------------------------------------------------
// Kernel 2: per-batch stable pos/neg compaction into T slots.
// ---------------------------------------------------------------------------
__global__ void k_select(const int* __restrict__ flag, int* __restrict__ src,
                         int* __restrict__ ispos) {
    int b = blockIdx.x;
    if (threadIdx.x != 0) return;
    const int* f = flag + b * NP;
    int pt = 0, nt = 0;
    for (int i = 0; i < NP; ++i) { pt += (f[i] == 1); nt += (f[i] == 2); }
    int pc = imin(pt, PMAX);
    // neg_needed = floor(float32(pc)/0.33f) - pc   (match JAX float32 math)
    int need = (int)floorf((float)pc / 0.33f) - pc;
    int nc = imin(imin(need, nt), TT - pc);
    nc = imax(nc, 0);
    int* s  = src + b * TT;
    int* ip = ispos + b * TT;
    for (int t = 0; t < TT; ++t) { s[t] = -1; ip[t] = 0; }
    int pw = 0, nw = 0;
    for (int i = 0; i < NP && (pw < pc || nw < nc); ++i) {
        if (f[i] == 1 && pw < pc)      { s[pw] = i; ip[pw] = 1; ++pw; }
        else if (f[i] == 2 && nw < nc) { s[pc + nw] = i; ++nw; }
    }
}

// ---------------------------------------------------------------------------
// Kernel 3: per-slot rois / cls / deltas + mm box for the mask stage.
// ---------------------------------------------------------------------------
__global__ void k_slots(const float* __restrict__ prop, const int* __restrict__ cls,
                        const float* __restrict__ gtb, const int* __restrict__ assign,
                        const int* __restrict__ src, const int* __restrict__ ispos,
                        float* __restrict__ out, float* __restrict__ mm_ws,
                        int* __restrict__ gidx_ws) {
    int idx = blockIdx.x * blockDim.x + threadIdx.x;
    if (idx >= BN * TT) return;
    int b = idx / TT;
    float* o_rois = out;
    float* o_cls  = out + (size_t)BN * TT * 4;
    float* o_del  = out + (size_t)BN * TT * 5;
    int s  = src[idx];
    int ip = ispos[idx];
    float r0 = 0, r1 = 0, r2 = 0, r3 = 0;
    float d0 = 0, d1 = 0, d2 = 0, d3 = 0, cv = 0;
    float m0 = 0, m1 = 0, m2 = 0, m3 = 0;
    int g = 0;
    if (s >= 0) {
        const float* p = prop + (size_t)(b * NP + s) * 4;
        r0 = p[0]; r1 = p[1]; r2 = p[2]; r3 = p[3];
        g = assign[b * NP + s];
        const float* q = gtb + (size_t)(b * NG + g) * 4;
        float gy1 = q[0], gx1 = q[1], gy2 = q[2], gx2 = q[3];
        if (ip) {
            cv = (float)cls[b * NG + g];
            float h  = fmaxf(r2 - r0, EPSF), w  = fmaxf(r3 - r1, EPSF);
            float cy = r0 + 0.5f * h,        cx = r1 + 0.5f * w;
            float gh = fmaxf(gy2 - gy1, EPSF), gw = fmaxf(gx2 - gx1, EPSF);
            float gcy = gy1 + 0.5f * gh,       gcx = gx1 + 0.5f * gw;
            d0 = ((gcy - cy) / h) * 10.f;         // /0.1
            d1 = ((gcx - cx) / w) * 10.f;
            d2 = logf(gh / h) * 5.f;              // /0.2
            d3 = logf(gw / w) * 5.f;
            m0 = (r0 - gy1) / gh; m1 = (r1 - gx1) / gw;
            m2 = (r2 - gy1) / gh; m3 = (r3 - gx1) / gw;
        }
    }
    o_rois[idx * 4 + 0] = r0; o_rois[idx * 4 + 1] = r1;
    o_rois[idx * 4 + 2] = r2; o_rois[idx * 4 + 3] = r3;
    o_cls[idx] = cv;
    o_del[idx * 4 + 0] = d0; o_del[idx * 4 + 1] = d1;
    o_del[idx * 4 + 2] = d2; o_del[idx * 4 + 3] = d3;
    mm_ws[idx * 4 + 0] = m0; mm_ws[idx * 4 + 1] = m1;
    mm_ws[idx * 4 + 2] = m2; mm_ws[idx * 4 + 3] = m3;
    gidx_ws[idx] = g;
}

// ---------------------------------------------------------------------------
// Fragment loader: 16 K-contiguous f16 elements as two 16B LDS loads.
// Layouts store the K axis contiguously, so lane L's elements
// {kb0..kb0+7, kb1..kb1+7} (+kt*32) are two aligned h8 vectors.
// ---------------------------------------------------------------------------
__device__ __forceinline__ v16h frag_ld(const _Float16* base, int row, int kt,
                                        int kb0, int kb1) {
    const h8 lo = *(const h8*)(base + row * 64 + kt * 32 + kb0);
    const h8 hi = *(const h8*)(base + row * 64 + kt * 32 + kb1);
    return __builtin_shufflevector(lo, hi, 0, 1, 2, 3, 4, 5, 6, 7,
                                           8, 9, 10, 11, 12, 13, 14, 15);
}

// ---------------------------------------------------------------------------
// Kernel 4: WMMA bilinear crop-resize, one wave32 per ROI.
//   out(28x28) = round( Wy(28x56) * M(56x56) * Wx^T(56x28) )
// v_wmma_f32_16x16x32_f16 tiles: stage1 2x4x2 = 16, stage2 2x2x2 = 8.
// All pads are exact zeros so every fragment load is unconditional.
// ---------------------------------------------------------------------------
__global__ __launch_bounds__(32) void k_mask(const float* __restrict__ masks,
        const float* __restrict__ mm_ws, const int* __restrict__ gidx_ws,
        const int* __restrict__ ispos, float* __restrict__ out_masks) {
    int slot = blockIdx.x;               // b*TT + t
    int b = slot / TT;
    int L = threadIdx.x;                 // lane 0..31
    float* O = out_masks + (size_t)slot * (MOx * MOx);

    if (!ispos[slot]) {                  // block-uniform branch
        for (int i = L; i < MOx * MOx; i += 32) O[i] = 0.f;
        return;
    }

    __shared__ __align__(16) _Float16 ldsMT[64 * 64];  // mask^T: [x][r], zero pad
    __shared__ __align__(16) _Float16 ldsWy[32 * 64];  // [i][r]
    __shared__ __align__(16) _Float16 ldsWx[32 * 64];  // [j][x]
    __shared__ __align__(16) _Float16 ldsS [32 * 64];  // stage-1: [i][x]

    // Zero-fill everything with 16B stores (same-wave LDS ops stay in order,
    // plus an explicit barrier before the scatter fills).
    {
        h8 z = {};
        h8* mt16 = (h8*)ldsMT;
        h8* wy16 = (h8*)ldsWy;
        h8* wx16 = (h8*)ldsWx;
        h8* s16  = (h8*)ldsS;
        for (int i = L; i < 64 * 64 / 8; i += 32) mt16[i] = z;
        for (int i = L; i < 32 * 64 / 8; i += 32) { wy16[i] = z; wx16[i] = z; s16[i] = z; }
    }
    __syncthreads();

    int g = gidx_ws[slot];
    float y1 = mm_ws[slot * 4 + 0], x1 = mm_ws[slot * 4 + 1];
    float y2 = mm_ws[slot * 4 + 2], x2 = mm_ws[slot * 4 + 3];

    // Gather the assigned mask (stride NG floats) into transposed f16 LDS.
    const float* srcm = masks + (size_t)b * (HMS * HMS * NG) + g;
    __builtin_prefetch(srcm, 0, 1);                      // global_prefetch_b8
    for (int i = L; i < HMS * HMS; i += 32) {
        int r = i / HMS;
        int x = i - r * HMS;
        ldsMT[x * 64 + r] = (_Float16)srcm[(size_t)i * NG];
    }

    // Materialize bilinear tap weights: lane L handles output row/col L (<28).
    if (L < MOx) {
        float ys = y1 * 55.f + (float)L * ((y2 - y1) * 55.f / 27.f);
        float fy = floorf(ys);
        float wy = ys - fy;
        int p0 = imin(imax((int)fy, 0), 55);
        int p1 = imin(imax((int)fy + 1, 0), 55);
        if (ys >= 0.f && ys <= 55.f) {
            ldsWy[L * 64 + p0] = (_Float16)(1.f - wy);
            _Float16 t = ldsWy[L * 64 + p1];             // handles p0==p1 (sum = 1)
            ldsWy[L * 64 + p1] = t + (_Float16)wy;
        }
        float xs = x1 * 55.f + (float)L * ((x2 - x1) * 55.f / 27.f);
        float fx = floorf(xs);
        float wx = xs - fx;
        int q0 = imin(imax((int)fx, 0), 55);
        int q1 = imin(imax((int)fx + 1, 0), 55);
        if (xs >= 0.f && xs <= 55.f) {
            ldsWx[L * 64 + q0] = (_Float16)(1.f - wx);
            _Float16 t = ldsWx[L * 64 + q1];
            ldsWx[L * 64 + q1] = t + (_Float16)wx;
        }
    }
    __syncthreads();

    const int lrow = L & 15;
    const int kb0 = (L < 16) ? 0 : 8;
    const int kb1 = (L < 16) ? 16 : 24;
    const int mofs = (L < 16) ? 0 : 8;

    // ---------------- stage 1: S = Wy * M  (A=Wy rows, B=ldsMT cols) --------
    #pragma unroll
    for (int mt = 0; mt < 2; ++mt) {
        int i = mt * 16 + lrow;
        v16h A0 = frag_ld(ldsWy, i, 0, kb0, kb1);
        v16h A1 = frag_ld(ldsWy, i, 1, kb0, kb1);
        #pragma unroll
        for (int nt = 0; nt < 4; ++nt) {
            int x = nt * 16 + lrow;
            v16h B0 = frag_ld(ldsMT, x, 0, kb0, kb1);
            v16h B1 = frag_ld(ldsMT, x, 1, kb0, kb1);
            v8f acc = {};
            acc = __builtin_amdgcn_wmma_f32_16x16x32_f16(false, A0, false, B0,
                                                         (short)0, acc, false, false);
            acc = __builtin_amdgcn_wmma_f32_16x16x32_f16(false, A1, false, B1,
                                                         (short)0, acc, false, false);
            int ibase = mt * 16 + mofs;
            #pragma unroll
            for (int v = 0; v < 8; ++v)
                ldsS[(ibase + v) * 64 + x] = (_Float16)acc[v];   // unconditional
        }
    }
    __syncthreads();

    // ---------------- stage 2: O = S * Wx^T  (A=ldsS rows, B=ldsWx rows) ----
    #pragma unroll
    for (int mt = 0; mt < 2; ++mt) {
        int i = mt * 16 + lrow;
        v16h A0 = frag_ld(ldsS, i, 0, kb0, kb1);
        v16h A1 = frag_ld(ldsS, i, 1, kb0, kb1);
        #pragma unroll
        for (int nt = 0; nt < 2; ++nt) {
            int j = nt * 16 + lrow;
            v16h B0 = frag_ld(ldsWx, j, 0, kb0, kb1);
            v16h B1 = frag_ld(ldsWx, j, 1, kb0, kb1);
            v8f acc = {};
            acc = __builtin_amdgcn_wmma_f32_16x16x32_f16(false, A0, false, B0,
                                                         (short)0, acc, false, false);
            acc = __builtin_amdgcn_wmma_f32_16x16x32_f16(false, A1, false, B1,
                                                         (short)0, acc, false, false);
            int ibase = mt * 16 + mofs;
            #pragma unroll
            for (int v = 0; v < 8; ++v) {
                int i2 = ibase + v;
                if (i2 < MOx && j < MOx)
                    O[i2 * MOx + j] = rintf(acc[v]);   // round-half-even == jnp.round
            }
        }
    }
}

// ---------------------------------------------------------------------------
extern "C" void kernel_launch(void* const* d_in, const int* in_sizes, int n_in,
                              void* d_out, int out_size, void* d_ws, size_t ws_size,
                              hipStream_t stream) {
    const float* prop = (const float*)d_in[0];   // [B,N,4]
    const int*   cls  = (const int*)d_in[1];     // [B,G]
    const float* gtb  = (const float*)d_in[2];   // [B,G,4]
    const float* msk  = (const float*)d_in[3];   // [B,56,56,G]
    float* out = (float*)d_out;

    int* wsI       = (int*)d_ws;
    int* ws_assign = wsI;                         // B*N
    int* ws_flag   = wsI + BN * NP;               // B*N
    int* ws_src    = wsI + 2 * BN * NP;           // B*T
    int* ws_ispos  = ws_src + BN * TT;            // B*T
    int* ws_gidx   = ws_ispos + BN * TT;          // B*T
    float* ws_mm   = (float*)(ws_gidx + BN * TT); // B*T*4

    k_iou<<<(BN * NP + 255) / 256, 256, 0, stream>>>(prop, cls, gtb, ws_assign, ws_flag);
    k_select<<<BN, 32, 0, stream>>>(ws_flag, ws_src, ws_ispos);
    k_slots<<<(BN * TT + 255) / 256, 256, 0, stream>>>(prop, cls, gtb, ws_assign,
                                                       ws_src, ws_ispos, out, ws_mm, ws_gidx);
    k_mask<<<BN * TT, 32, 0, stream>>>(msk, ws_mm, ws_gidx, ws_ispos,
                                       out + (size_t)BN * TT * 9);
}
